// JointWassersteinDistance_89446988906828
// MI455X (gfx1250) — matrix operand order repaired and checked by reading, not compiled, not run on hardware
//
#include <hip/hip_runtime.h>
#include <hip/hip_bf16.h>

typedef _Float16 v16h __attribute__((ext_vector_type(16)));
typedef float    v8f  __attribute__((ext_vector_type(8)));
typedef _Float16 h8   __attribute__((ext_vector_type(8)));
typedef _Float16 h4   __attribute__((ext_vector_type(4)));

#define NROW 8192
#define NCOL 8192
#define DIM  512
#define NCLS 10
#define REG_E 0.05f
#define NEGREG (-20.0f)               // -1/REG_E
#define N_ITER 20
#define LOGAB (-9.010913347279288f)   // log(1/8192)

union FragU { v16h v; uint4 q[2]; };

// CDNA5 async global -> LDS copy (16B per lane), tracked by ASYNCcnt.
__device__ __forceinline__ void async_load_b128(unsigned lds_byte_off, const void* gaddr) {
    asm volatile("global_load_async_to_lds_b128 %0, %1, off"
                 :: "v"(lds_byte_off), "v"(gaddr) : "memory");
}
#define WAIT_ASYNC(n) asm volatile("s_wait_asynccnt " #n ::: "memory")

__device__ __forceinline__ unsigned lds_off(const void* p) {
    return (unsigned)(size_t)p;   // low 32 bits of flat addr == LDS byte offset
}

// ---------------------------------------------------------------- init
__global__ void init_kernel(float* u, float* v, float* scal) {
    int idx = blockIdx.x * 256 + threadIdx.x;
    if (idx < NROW) { u[idx] = 0.0f; v[idx] = 0.0f; }
    if (idx < 4) scal[idx] = 0.0f;
}

// ---------------------------------------------------------------- fp32 -> fp16
__global__ void convert_kernel(const float* __restrict__ X, _Float16* __restrict__ Xh, int n) {
    for (int idx = blockIdx.x * blockDim.x + threadIdx.x; idx < n; idx += gridDim.x * blockDim.x)
        Xh[idx] = (_Float16)X[idx];
}

// ---------------------------------------------------------------- row norms
__global__ __launch_bounds__(128)
void norms_kernel(const float* __restrict__ X, float* __restrict__ out, int dim) {
    __shared__ float red[128];
    const int i = blockIdx.x, t = threadIdx.x;
    float s = 0.0f;
    for (int d = t; d < dim; d += 128) { float v = X[(size_t)i * dim + d]; s = fmaf(v, v, s); }
    red[t] = s; __syncthreads();
    for (int o = 64; o > 0; o >>= 1) { if (t < o) red[t] += red[t + o]; __syncthreads(); }
    if (t == 0) out[i] = red[0];
}

// ---------------------------------------------------------------- WMMA GEMM: CX + maxes
// 64x128 macro-tile, 8 waves each owning a 32x32 region (2x2 WMMA tiles),
// double-buffered LDS staged via global_load_async_to_lds_b128.
__global__ __launch_bounds__(256)
void gemm_cx_kernel(const _Float16* __restrict__ XPh, const _Float16* __restrict__ XQh,
                    const float* __restrict__ a2, const float* __restrict__ b2,
                    const float* __restrict__ YP, const float* __restrict__ YQ,
                    float* __restrict__ Cbuf, float* __restrict__ scal)
{
    __shared__ __align__(16) _Float16 As[2][64][32];    // 2 x 4KB
    __shared__ __align__(16) _Float16 Bs[2][128][32];   // 2 x 8KB
    __shared__ float red[256];

    const int t    = threadIdx.x;
    const int lane = t & 31, wv = t >> 5;
    const int wr   = wv >> 2, wc = wv & 3;
    const int lm   = lane & 15, kg = lane >> 4;
    const int row0 = blockIdx.y * 64;
    const int col0 = blockIdx.x * 128;

    v8f acc[2][2];
    for (int i = 0; i < 2; ++i)
        for (int j = 0; j < 2; ++j)
            for (int e = 0; e < 8; ++e) acc[i][j][e] = 0.0f;

    const int ar = t >> 2, ac = (t & 3) * 8;   // A tile: 64 rows x 32 halves, 16B/thread
    const int bn = t >> 1, bc = (t & 1) * 16;  // B tile: 128 rows x 32 halves, 32B/thread
    const _Float16* gA = XPh + (size_t)(row0 + ar) * DIM + ac;
    const _Float16* gB = XQh + (size_t)(col0 + bn) * DIM + bc;

    // prologue: stage chunk 0 into buffer 0 (3 async ops per thread)
    async_load_b128(lds_off(&As[0][ar][ac]), gA);
    async_load_b128(lds_off(&Bs[0][bn][bc]), gB);
    async_load_b128(lds_off(&Bs[0][bn][bc + 8]), gB + 8);

    for (int kc = 0; kc < DIM / 32; ++kc) {
        const int b = kc & 1;
        if (kc + 1 < DIM / 32) {
            const int kb = (kc + 1) * 32;
            async_load_b128(lds_off(&As[b ^ 1][ar][ac]), gA + kb);
            async_load_b128(lds_off(&Bs[b ^ 1][bn][bc]), gB + kb);
            async_load_b128(lds_off(&Bs[b ^ 1][bn][bc + 8]), gB + kb + 8);
            WAIT_ASYNC(3);   // in-order completion => current chunk's 3 ops done
        } else {
            WAIT_ASYNC(0);
        }
        __syncthreads();     // all waves' async data visible in LDS

        FragU af[2], bf[2];
        // A 16x32 fp16 layout: lane m=lane&15, kg=lane>>4; K = 16*(h>>3) + 8*kg + (h&7)
        for (int ti = 0; ti < 2; ++ti) {
            const int r = wr * 32 + ti * 16 + lm;
            af[ti].q[0] = *(const uint4*)&As[b][r][kg * 8];
            af[ti].q[1] = *(const uint4*)&As[b][r][16 + kg * 8];
        }
        // B 32x16 fp16 layout: lane n=lane&15, kg=lane>>4; K = 16*(h>>3) + 8*kg + (h&7)
        for (int tj = 0; tj < 2; ++tj) {
            const int n = wc * 32 + tj * 16 + lm;
            bf[tj].q[0] = *(const uint4*)&Bs[b][n][kg * 8];
            bf[tj].q[1] = *(const uint4*)&Bs[b][n][16 + kg * 8];
        }
        for (int ti = 0; ti < 2; ++ti)
            for (int tj = 0; tj < 2; ++tj)
                acc[ti][tj] = __builtin_amdgcn_wmma_f32_16x16x32_f16(
                    false, af[ti].v, false, bf[tj].v, (short)0, acc[ti][tj], false, false);
        __syncthreads();     // done reading buf b before it is overwritten next iter
    }

    // Epilogue: cx = a2 + b2 - 2*dot (clamped), fused label-distance max tracking.
    // C/D layout: lanes 0-15 -> M=r, N=lane; lanes 16-31 -> M=8+r, N=lane-16.
    float mcx = 0.0f, mcy = 0.0f;
    for (int tj = 0; tj < 2; ++tj) {
        const int gj = col0 + wc * 32 + tj * 16 + lm;
        float yq[NCLS];
        for (int d = 0; d < NCLS; ++d) yq[d] = YQ[gj * NCLS + d];
        const float b2v = b2[gj];
        for (int ti = 0; ti < 2; ++ti) {
            const int gi0 = row0 + wr * 32 + ti * 16 + kg * 8;
            for (int r = 0; r < 8; ++r) {
                const int gi = gi0 + r;
                const float cx = fmaxf(a2[gi] + b2v - 2.0f * acc[ti][tj][r], 0.0f);
                float cy = 0.0f;
                for (int d = 0; d < NCLS; ++d) {
                    const float df = YP[gi * NCLS + d] - yq[d];
                    cy = fmaf(df, df, cy);
                }
                Cbuf[(size_t)gi * NCOL + gj] = cx;
                mcx = fmaxf(mcx, cx);
                mcy = fmaxf(mcy, cy);
            }
        }
    }
    red[t] = mcx; __syncthreads();
    for (int o = 128; o > 0; o >>= 1) { if (t < o) red[t] = fmaxf(red[t], red[t + o]); __syncthreads(); }
    if (t == 0) atomicMax((int*)&scal[0], __float_as_int(red[0]));
    __syncthreads();
    red[t] = mcy; __syncthreads();
    for (int o = 128; o > 0; o >>= 1) { if (t < o) red[t] = fmaxf(red[t], red[t + o]); __syncthreads(); }
    if (t == 0) atomicMax((int*)&scal[1], __float_as_int(red[0]));
}

// ---------------------------------------------------------------- C = CX + w*CY, cmax
__global__ __launch_bounds__(256)
void combine_kernel(const float* __restrict__ YP, const float* __restrict__ YQ,
                    float* __restrict__ Cbuf, float* __restrict__ scal, int total4)
{
    __shared__ float red[256];
    const float cxm = scal[0], cym = scal[1];
    const float w = (cym > 0.0f) ? (cxm / cym) : 0.0f;
    float4* C4 = (float4*)Cbuf;
    float mx = 0.0f;
    for (int idx = blockIdx.x * 256 + threadIdx.x; idx < total4; idx += gridDim.x * 256) {
        const int e0 = idx * 4;
        const int i = e0 >> 13, j0 = e0 & 8191;
        float4 c4 = C4[idx];
        float cc[4] = {c4.x, c4.y, c4.z, c4.w};
        for (int e = 0; e < 4; ++e) {
            float cy = 0.0f;
            for (int d = 0; d < NCLS; ++d) {
                const float df = YP[i * NCLS + d] - YQ[(j0 + e) * NCLS + d];
                cy = fmaf(df, df, cy);
            }
            cc[e] = fmaf(w, cy, cc[e]);
            mx = fmaxf(mx, cc[e]);
        }
        C4[idx] = make_float4(cc[0], cc[1], cc[2], cc[3]);
    }
    const int t = threadIdx.x;
    red[t] = mx; __syncthreads();
    for (int o = 128; o > 0; o >>= 1) { if (t < o) red[t] = fmaxf(red[t], red[t + o]); __syncthreads(); }
    if (t == 0) atomicMax((int*)&scal[2], __float_as_int(red[0]));
}

// ---------------------------------------------------------------- Kh = fp16(C / cmax): 128 MB, L2-resident
__global__ __launch_bounds__(256)
void normalize_kernel(const float* __restrict__ Cbuf, _Float16* __restrict__ Kh,
                      const float* __restrict__ scal, int total4)
{
    const float cm = scal[2];
    const float inv = (cm > 0.0f) ? (1.0f / cm) : 0.0f;
    const float4* C4 = (const float4*)Cbuf;
    h4* K4 = (h4*)Kh;
    for (int idx = blockIdx.x * 256 + threadIdx.x; idx < total4; idx += gridDim.x * 256) {
        const float4 c = C4[idx];
        h4 hv;
        hv[0] = (_Float16)(c.x * inv); hv[1] = (_Float16)(c.y * inv);
        hv[2] = (_Float16)(c.z * inv); hv[3] = (_Float16)(c.w * inv);
        K4[idx] = hv;
    }
}

// ---------------------------------------------------------------- u = log_a - lse_j(K + v), K = -Kh/reg
__global__ __launch_bounds__(256)
void u_update_kernel(const _Float16* __restrict__ Kh, const float* __restrict__ v,
                     float* __restrict__ u)
{
    __shared__ float sm[256], ss[256];
    const int i = blockIdx.x, t = threadIdx.x;
    const h8* row = (const h8*)(Kh + (size_t)i * NCOL);     // 1024 vectors of 8 halves
    const float4* v4 = (const float4*)v;
    float m = -__builtin_inff(), s = 0.0f;
    for (int c = t; c < NCOL / 8; c += 256) {
        const h8 hv = row[c];
        const float4 va = v4[c * 2], vb = v4[c * 2 + 1];
        float x[8];
        x[0] = fmaf((float)hv[0], NEGREG, va.x); x[1] = fmaf((float)hv[1], NEGREG, va.y);
        x[2] = fmaf((float)hv[2], NEGREG, va.z); x[3] = fmaf((float)hv[3], NEGREG, va.w);
        x[4] = fmaf((float)hv[4], NEGREG, vb.x); x[5] = fmaf((float)hv[5], NEGREG, vb.y);
        x[6] = fmaf((float)hv[6], NEGREG, vb.z); x[7] = fmaf((float)hv[7], NEGREG, vb.w);
        for (int e = 0; e < 8; ++e) {
            if (x[e] > m) { s = fmaf(s, __expf(m - x[e]), 1.0f); m = x[e]; }
            else          { s += __expf(x[e] - m); }
        }
    }
    sm[t] = m; ss[t] = s; __syncthreads();
    for (int o = 128; o > 0; o >>= 1) {
        if (t < o) {
            const float m1 = sm[t], m2 = sm[t + o], s1 = ss[t], s2 = ss[t + o];
            const float mm = fmaxf(m1, m2);
            sm[t] = mm;
            ss[t] = s1 * __expf(m1 - mm) + s2 * __expf(m2 - mm);
        }
        __syncthreads();
    }
    if (t == 0) u[i] = LOGAB - (sm[0] + __logf(ss[0]));
}

// ---------------------------------------------------------------- column lse partials (coalesced b128)
__global__ __launch_bounds__(256)
void v_partial_kernel(const _Float16* __restrict__ Kh, const float* __restrict__ u,
                      float* __restrict__ pm, float* __restrict__ ps)
{
    const int j0 = (blockIdx.x * 256 + threadIdx.x) * 8;   // 8 consecutive columns
    const int r0 = blockIdx.y * 256;                        // row chunk
    float m8[8], s8[8];
    for (int e = 0; e < 8; ++e) { m8[e] = -__builtin_inff(); s8[e] = 0.0f; }
    for (int r = r0; r < r0 + 256; ++r) {
        const h8 hv = *(const h8*)(Kh + (size_t)r * NCOL + j0);
        const float ur = u[r];
        for (int e = 0; e < 8; ++e) {
            const float x = fmaf((float)hv[e], NEGREG, ur);
            if (x > m8[e]) { s8[e] = fmaf(s8[e], __expf(m8[e] - x), 1.0f); m8[e] = x; }
            else           { s8[e] += __expf(x - m8[e]); }
        }
    }
    for (int e = 0; e < 8; ++e) {
        pm[blockIdx.y * NCOL + j0 + e] = m8[e];
        ps[blockIdx.y * NCOL + j0 + e] = s8[e];
    }
}

// ---------------------------------------------------------------- merge partials -> v
__global__ __launch_bounds__(256)
void v_final_kernel(const float* __restrict__ pm, const float* __restrict__ ps,
                    float* __restrict__ v)
{
    const int j = blockIdx.x * 256 + threadIdx.x;
    float m = pm[j], s = ps[j];
    for (int c = 1; c < 32; ++c) {
        const float m2 = pm[c * NCOL + j], s2 = ps[c * NCOL + j];
        const float mm = fmaxf(m, m2);
        s = s * __expf(m - mm) + s2 * __expf(m2 - mm);
        m = mm;
    }
    v[j] = LOGAB - (m + __logf(s));
}

// ---------------------------------------------------------------- sum(C * exp(-C/(reg*cmax)+u+v)) per row
__global__ __launch_bounds__(256)
void obj_row_kernel(const float* __restrict__ Cbuf, const float* __restrict__ u,
                    const float* __restrict__ v, const float* __restrict__ scal,
                    float* __restrict__ rowpart)
{
    __shared__ float red[256];
    const int i = blockIdx.x, t = threadIdx.x;
    const float negInv = -1.0f / (REG_E * scal[2]);
    const float ui = u[i];
    const float4* row4 = (const float4*)(Cbuf + (size_t)i * NCOL);
    const float4* v4 = (const float4*)v;
    float s = 0.0f;
    for (int c = t; c < NCOL / 4; c += 256) {
        const float4 c4 = row4[c];
        const float4 vv = v4[c];
        s = fmaf(c4.x, __expf(fmaf(c4.x, negInv, ui + vv.x)), s);
        s = fmaf(c4.y, __expf(fmaf(c4.y, negInv, ui + vv.y)), s);
        s = fmaf(c4.z, __expf(fmaf(c4.z, negInv, ui + vv.z)), s);
        s = fmaf(c4.w, __expf(fmaf(c4.w, negInv, ui + vv.w)), s);
    }
    red[t] = s; __syncthreads();
    for (int o = 128; o > 0; o >>= 1) { if (t < o) red[t] += red[t + o]; __syncthreads(); }
    if (t == 0) rowpart[i] = red[0];
}

__global__ __launch_bounds__(256)
void obj_final_kernel(const float* __restrict__ rowpart, float* __restrict__ out)
{
    __shared__ float red[256];
    const int t = threadIdx.x;
    float s = 0.0f;
    for (int i = t; i < NROW; i += 256) s += rowpart[i];
    red[t] = s; __syncthreads();
    for (int o = 128; o > 0; o >>= 1) { if (t < o) red[t] += red[t + o]; __syncthreads(); }
    if (t == 0) out[0] = red[0];
}

// ================================================================ launch
extern "C" void kernel_launch(void* const* d_in, const int* in_sizes, int n_in,
                              void* d_out, int out_size, void* d_ws, size_t ws_size,
                              hipStream_t stream)
{
    const float* XQ = (const float*)d_in[0];   // (8192, 512)
    const float* YQ = (const float*)d_in[1];   // (8192, 10)
    const float* XP = (const float*)d_in[2];   // (8192, 512)
    const float* YP = (const float*)d_in[3];   // (8192, 10)

    char* w = (char*)d_ws;
    float*    Cbuf = (float*)w;     w += (size_t)NROW * NCOL * sizeof(float);      // 256 MB
    _Float16* Kh   = (_Float16*)w;  w += (size_t)NROW * NCOL * sizeof(_Float16);   // 128 MB (L2-resident)
    _Float16* XPh  = (_Float16*)w;  w += (size_t)NROW * DIM * sizeof(_Float16);    // 8 MB
    _Float16* XQh  = (_Float16*)w;  w += (size_t)NCOL * DIM * sizeof(_Float16);    // 8 MB
    float*    a2   = (float*)w;     w += NROW * sizeof(float);
    float*    b2   = (float*)w;     w += NCOL * sizeof(float);
    float*    u    = (float*)w;     w += NROW * sizeof(float);
    float*    v    = (float*)w;     w += NCOL * sizeof(float);
    float*    pm   = (float*)w;     w += (size_t)32 * NCOL * sizeof(float);        // 1 MB
    float*    ps   = (float*)w;     w += (size_t)32 * NCOL * sizeof(float);        // 1 MB
    float*    rowp = (float*)w;     w += NROW * sizeof(float);
    float*    scal = (float*)w;     w += 16 * sizeof(float);   // [0]=cxmax [1]=cymax [2]=cmax

    const int TOT4 = NROW * NCOL / 4;   // 16,777,216 float4s

    init_kernel<<<32, 256, 0, stream>>>(u, v, scal);
    convert_kernel<<<2048, 256, 0, stream>>>(XP, XPh, NROW * DIM);
    convert_kernel<<<2048, 256, 0, stream>>>(XQ, XQh, NCOL * DIM);
    norms_kernel<<<NROW, 128, 0, stream>>>(XP, a2, DIM);
    norms_kernel<<<NCOL, 128, 0, stream>>>(XQ, b2, DIM);

    gemm_cx_kernel<<<dim3(NCOL / 128, NROW / 64), 256, 0, stream>>>(
        XPh, XQh, a2, b2, YP, YQ, Cbuf, scal);

    combine_kernel<<<16384, 256, 0, stream>>>(YP, YQ, Cbuf, scal, TOT4);
    normalize_kernel<<<16384, 256, 0, stream>>>(Cbuf, Kh, scal, TOT4);

    for (int it = 0; it < N_ITER; ++it) {
        u_update_kernel<<<NROW, 256, 0, stream>>>(Kh, v, u);
        v_partial_kernel<<<dim3(NCOL / 2048, NROW / 256), 256, 0, stream>>>(Kh, u, pm, ps);
        v_final_kernel<<<NCOL / 256, 256, 0, stream>>>(pm, ps, v);
    }

    obj_row_kernel<<<NROW, 256, 0, stream>>>(Cbuf, u, v, scal, rowp);
    obj_final_kernel<<<1, 256, 0, stream>>>(rowp, (float*)d_out);
}
